// mylstm_63591285785026
// MI455X (gfx1250) — matrix operand order repaired and checked by reading, not compiled, not run on hardware
//
#include <hip/hip_runtime.h>
#include <hip/hip_bf16.h>
#include <math.h>

typedef __attribute__((ext_vector_type(16))) __bf16 bfv16;
typedef __attribute__((ext_vector_type(8)))  float  f32v8;

#define T_    240
#define B_    240
#define INP_  256
#define HID_  1024
#define G4H   4096
#define KTOT  1280
#define MT    15     // 240/16 row tiles
#define KT_X  8      // 256/32 k-tiles from x
#define KT_H  32     // 1024/32 k-tiles from h
#define KT_ALL 40
#define NSTEPS_ 30
#define OUT_  8
#define NBLK  60     // persistent encoder grid (60 blocks x 8 waves = 480 waves)

__device__ __forceinline__ float sigf(float x) { return 1.0f / (1.0f + __expf(-x)); }

// device-wide barrier: monotone counter, barrier b complete when cnt == b*NBLK
__device__ __forceinline__ void gsync(unsigned* cnt, unsigned b) {
    __syncthreads();
    if (threadIdx.x == 0) {
        __threadfence();
        atomicAdd(cnt, 1u);
        while (*(volatile unsigned*)cnt < b * NBLK) {
            __builtin_amdgcn_s_sleep(1);
        }
    }
    __syncthreads();
}

// ---------------- init: zero recurrent state + barrier counter ----------------
__global__ void k_init(float* c, float* hf, __bf16* hp, unsigned* bar) {
    int i = blockIdx.x * blockDim.x + threadIdx.x;
    if (i == 0) *bar = 0u;
    if (i < B_ * HID_) {
        c[i]  = 0.0f;
        hf[i] = 0.0f;
        hp[i] = (__bf16)0.0f;
    }
}

// ---------------- pack W (4096 x 1280) into bf16 B-fragment layout ----------------
__global__ void k_pack_w(const float* __restrict__ wih, const float* __restrict__ whh,
                         __bf16* __restrict__ wp) {
    int i = blockIdx.x * blockDim.x + threadIdx.x;
    if (i >= G4H * KTOT) return;
    int n = i / KTOT, k = i % KTOT;
    float w = (k < INP_) ? wih[(size_t)n * INP_ + k] : whh[(size_t)n * HID_ + (k - INP_)];
    int nt = n >> 4, nl = n & 15, kt = k >> 5, ki = k & 31;
    int lane = nl + (((ki >> 3) & 1) ? 16 : 0);
    int sub  = (ki & 7) + ((ki & 16) ? 8 : 0);
    wp[((size_t)(nt * KT_ALL + kt) * 32 + lane) * 16 + sub] = (__bf16)w;
}

// ---------------- pack all X_t into bf16 A-fragment layout ----------------
__global__ void k_pack_x(const float* __restrict__ data, __bf16* __restrict__ xp) {
    long long i = (long long)blockIdx.x * blockDim.x + threadIdx.x;
    if (i >= (long long)T_ * B_ * INP_) return;
    int k = (int)(i % INP_);
    int m = (int)((i / INP_) % B_);
    int t = (int)(i / ((long long)INP_ * B_));
    float x = data[i];
    int mt = m >> 4, ml = m & 15, kt = k >> 5, ki = k & 31;
    int lane = ml + (((ki >> 3) & 1) ? 16 : 0);
    int sub  = (ki & 7) + ((ki & 16) ? 8 : 0);
    xp[(((size_t)t * MT + mt) * KT_X + kt) * 32 * 16 + (size_t)lane * 16 + sub] = (__bf16)x;
}

// ---------------- persistent fused encoder: 240 steps of (WMMA GEMM -> cell) ----------------
__global__ __launch_bounds__(256) void k_encoder(
    const bfv16* __restrict__ xpack,   // [T][MT][KT_X][32] fragments
    const bfv16* __restrict__ wp,      // [NT][KT_ALL][32] fragments
    __bf16* hpraw,                     // h bf16 A-fragments [MT][KT_H][32][16]
    const float* __restrict__ bih, const float* __restrict__ bhh,
    float* __restrict__ G, float* __restrict__ c, float* __restrict__ hf,
    unsigned* bar) {
    const int tid  = threadIdx.x;
    const int gtid = blockIdx.x * 256 + tid;
    const int wid  = gtid >> 5;
    const int lane = tid & 31;
    const int mt = wid >> 5;   // 0..14
    const int ng = wid & 31;   // 0..31 (8 n-tiles each)
    unsigned nbar = 0;

    const bfv16* hpv = (const bfv16*)hpraw;
    const bfv16* ah = hpv + (size_t)mt * KT_H * 32 + lane;
    const bfv16* wb = wp  + (size_t)ng * 8 * KT_ALL * 32 + lane;
    const int col0 = (ng * 8) * 16 + (lane & 15);
    const int row0 = mt * 16 + (lane >> 4) * 8;

    for (int t = 0; t < T_; ++t) {
        const bfv16* ax = xpack + ((size_t)t * MT + mt) * KT_X * 32 + lane;

        f32v8 acc[8];
#pragma unroll
        for (int n = 0; n < 8; ++n) acc[n] = (f32v8){0, 0, 0, 0, 0, 0, 0, 0};

        // software pipeline: double-buffered A/B fragments in registers
        bfv16 ab[2];
        bfv16 bb[2][8];
        ab[0] = ax[0];
#pragma unroll
        for (int n = 0; n < 8; ++n) bb[0][n] = wb[(n * KT_ALL) * 32];

#pragma unroll 2
        for (int kt = 0; kt < KT_ALL; ++kt) {
            const int cur = kt & 1, nxt = cur ^ 1;
            if (kt + 1 < KT_ALL) {
                ab[nxt] = (kt + 1 < KT_X) ? ax[(kt + 1) * 32] : ah[(kt + 1 - KT_X) * 32];
#pragma unroll
                for (int n = 0; n < 8; ++n)
                    bb[nxt][n] = wb[(n * KT_ALL + kt + 1) * 32];
            }
#pragma unroll
            for (int n = 0; n < 8; ++n)
                acc[n] = __builtin_amdgcn_wmma_f32_16x16x32_bf16(
                    false, ab[cur], false, bb[cur][n], (short)0, acc[n], false, false);
        }

        // store gate pre-activations (C/D layout -> row-major G)
#pragma unroll
        for (int n = 0; n < 8; ++n) {
            int col = col0 + n * 16;
#pragma unroll
            for (int v = 0; v < 8; ++v)
                G[(size_t)(row0 + v) * G4H + col] = acc[n][v];
        }

        gsync(bar, ++nbar);  // G complete

        // LSTM cell: 245760 elements over 15360 threads -> 16 each
#pragma unroll 1
        for (int r = 0; r < 16; ++r) {
            int i = gtid + r * (NBLK * 256);
            int m = i >> 10, j = i & 1023;
            const float* gr = G + (size_t)m * G4H;
            float gi = gr[j]        + bih[j]        + bhh[j];
            float gf = gr[1024 + j] + bih[1024 + j] + bhh[1024 + j];
            float gg = gr[2048 + j] + bih[2048 + j] + bhh[2048 + j];
            float go = gr[3072 + j] + bih[3072 + j] + bhh[3072 + j];
            float cc = sigf(gf) * c[i] + sigf(gi) * tanhf(gg);
            c[i] = cc;
            float h = sigf(go) * tanhf(cc);
            hf[i] = h;
            int hmt = m >> 4, ml = m & 15, hkt = j >> 5, ki = j & 31;
            int hlane = ml + (((ki >> 3) & 1) ? 16 : 0);
            int sub   = (ki & 7) + ((ki & 16) ? 8 : 0);
            hpraw[((size_t)(hmt * KT_H + hkt) * 32 + hlane) * 16 + sub] = (__bf16)h;
        }

        gsync(bar, ++nbar);  // h pack complete before next step's GEMM
    }
}

// ---------------- decoder: 30 serial attention+GRU steps, one persistent block ----------------
__global__ __launch_bounds__(1024) void k_decoder(
    const float* __restrict__ ctx,
    const float* __restrict__ attn_W, const float* __restrict__ attn_b,
    const float* __restrict__ comb_W, const float* __restrict__ comb_b,
    const float* __restrict__ gWih, const float* __restrict__ gWhh,
    const float* __restrict__ gbih, const float* __restrict__ gbhh,
    const float* __restrict__ outW, const float* __restrict__ outb,
    float* __restrict__ out) {
    __shared__ float su[OUT_];
    __shared__ float sh[HID_];
    __shared__ float saw[B_];
    __shared__ float sap[HID_];
    __shared__ float sco[HID_];
    __shared__ float sred[2];
    __shared__ float slog[OUT_];

    int tid = threadIdx.x;
    if (tid < OUT_) su[tid] = 1.0f / OUT_;
    sh[tid] = ctx[(size_t)(B_ - 1) * HID_ + tid];
    __syncthreads();

    for (int s = 0; s < NSTEPS_; ++s) {
        if (tid < B_) {
            const float* wr = attn_W + (size_t)tid * (HID_ + OUT_);
            float e = attn_b[tid];
            for (int q = 0; q < OUT_; ++q) e += su[q] * wr[q];
            for (int j = 0; j < HID_; ++j) e += sh[j] * wr[OUT_ + j];
            saw[tid] = e;
        }
        __syncthreads();
        if (tid == 0) {
            float mx = saw[0];
            for (int r = 1; r < B_; ++r) mx = fmaxf(mx, saw[r]);
            float sum = 0.0f;
            for (int r = 0; r < B_; ++r) { float v = __expf(saw[r] - mx); saw[r] = v; sum += v; }
            sred[0] = 1.0f / sum;
        }
        __syncthreads();
        if (tid < B_) saw[tid] *= sred[0];
        __syncthreads();
        {
            float a = 0.0f;
            for (int r = 0; r < B_; ++r) a += saw[r] * ctx[(size_t)r * HID_ + tid];
            sap[tid] = a;
        }
        __syncthreads();
        {
            const float* wr = comb_W + (size_t)tid * (HID_ + OUT_);
            float v = comb_b[tid];
            for (int q = 0; q < OUT_; ++q) v += su[q] * wr[q];
            for (int j = 0; j < HID_; ++j) v += sap[j] * wr[OUT_ + j];
            sco[tid] = fmaxf(v, 0.0f);
        }
        __syncthreads();
        float hnew;
        {
            float ir = gbih[tid], iz = gbih[1024 + tid], in_ = gbih[2048 + tid];
            float hr = gbhh[tid], hz = gbhh[1024 + tid], hn  = gbhh[2048 + tid];
            const float* wi0 = gWih + (size_t)tid * HID_;
            const float* wi1 = gWih + (size_t)(1024 + tid) * HID_;
            const float* wi2 = gWih + (size_t)(2048 + tid) * HID_;
            const float* wh0 = gWhh + (size_t)tid * HID_;
            const float* wh1 = gWhh + (size_t)(1024 + tid) * HID_;
            const float* wh2 = gWhh + (size_t)(2048 + tid) * HID_;
            for (int j = 0; j < HID_; ++j) {
                float cj = sco[j], hj = sh[j];
                ir += cj * wi0[j]; iz += cj * wi1[j]; in_ += cj * wi2[j];
                hr += hj * wh0[j]; hz += hj * wh1[j]; hn  += hj * wh2[j];
            }
            float r = sigf(ir + hr), z = sigf(iz + hz);
            float n = tanhf(in_ + r * hn);
            hnew = (1.0f - z) * n + z * sh[tid];
        }
        __syncthreads();
        sh[tid] = hnew;
        __syncthreads();
        if (tid < OUT_) {
            const float* wr = outW + (size_t)tid * HID_;
            float v = outb[tid];
            for (int j = 0; j < HID_; ++j) v += sh[j] * wr[j];
            slog[tid] = v;
        }
        __syncthreads();
        if (tid == 0) {
            float mx = slog[0];
            for (int o = 1; o < OUT_; ++o) mx = fmaxf(mx, slog[o]);
            float sum = 0.0f;
            for (int o = 0; o < OUT_; ++o) sum += __expf(slog[o] - mx);
            sred[1] = mx + __logf(sum);
        }
        __syncthreads();
        if (tid < OUT_) {
            float lp = slog[tid] - sred[1];
            out[s * OUT_ + tid] = lp;
            su[tid] = lp;
        }
        __syncthreads();
    }
}

// ---------------- host launch ----------------
extern "C" void kernel_launch(void* const* d_in, const int* in_sizes, int n_in,
                              void* d_out, int out_size, void* d_ws, size_t ws_size,
                              hipStream_t stream) {
    (void)in_sizes; (void)n_in; (void)out_size; (void)ws_size;
    const float* data    = (const float*)d_in[0];
    const float* enc_Wih = (const float*)d_in[1];
    const float* enc_Whh = (const float*)d_in[2];
    const float* enc_bih = (const float*)d_in[3];
    const float* enc_bhh = (const float*)d_in[4];
    const float* attn_W  = (const float*)d_in[5];
    const float* attn_b  = (const float*)d_in[6];
    const float* comb_W  = (const float*)d_in[7];
    const float* comb_b  = (const float*)d_in[8];
    const float* gru_Wih = (const float*)d_in[9];
    const float* gru_Whh = (const float*)d_in[10];
    const float* gru_bih = (const float*)d_in[11];
    const float* gru_bhh = (const float*)d_in[12];
    const float* out_W   = (const float*)d_in[13];
    const float* out_b   = (const float*)d_in[14];
    float* out = (float*)d_out;

    char* base = (char*)d_ws;
    size_t off = 0;
    auto take = [&](size_t bytes) { char* p = base + off; off = (off + bytes + 255) & ~(size_t)255; return p; };
    __bf16*   wpack = (__bf16*)take((size_t)G4H * KTOT * sizeof(__bf16));                 // 10.5 MB
    __bf16*   xpack = (__bf16*)take((size_t)T_ * MT * KT_X * 32 * 16 * sizeof(__bf16));   // 29.5 MB
    __bf16*   hpack = (__bf16*)take((size_t)MT * KT_H * 32 * 16 * sizeof(__bf16));        // 0.5 MB
    float*    G     = (float*)take((size_t)B_ * G4H * sizeof(float));                     // 3.9 MB
    float*    c     = (float*)take((size_t)B_ * HID_ * sizeof(float));                    // 1.0 MB
    float*    hf    = (float*)take((size_t)B_ * HID_ * sizeof(float));                    // 1.0 MB
    unsigned* bar   = (unsigned*)take(256);

    k_init<<<(B_ * HID_ + 255) / 256, 256, 0, stream>>>(c, hf, hpack, bar);
    k_pack_w<<<(G4H * KTOT + 255) / 256, 256, 0, stream>>>(enc_Wih, enc_Whh, wpack);
    {
        long long n = (long long)T_ * B_ * INP_;
        k_pack_x<<<(unsigned)((n + 255) / 256), 256, 0, stream>>>(data, xpack);
    }

    // single persistent kernel runs the full 240-step scan
    k_encoder<<<NBLK, 256, 0, stream>>>((const bfv16*)xpack, (const bfv16*)wpack, hpack,
                                        enc_bih, enc_bhh, G, c, hf, bar);

    k_decoder<<<1, 1024, 0, stream>>>(hf, attn_W, attn_b, comb_W, comb_b,
                                      gru_Wih, gru_Whh, gru_bih, gru_bhh,
                                      out_W, out_b, out);
}
